// SNU_Network_classification_16561393893855
// MI455X (gfx1250) — matrix operand order repaired and checked by reading, not compiled, not run on hardware
//
#include <hip/hip_runtime.h>
#include <hip/hip_bf16.h>

// ---------------------------------------------------------------------------
// SNU 4-layer spiking network forward pass for MI455X (gfx1250).
// Strategy: bf16 WMMA GEMMs (v_wmma_f32_16x16x32_bf16) with split-K for
// occupancy; weights converted fp32->bf16 once per call (L2-resident, 96MB);
// SNU nonlinearity fused in a partial-sum reduction epilogue.
// Workspace requirement: ~155 MB.
// ---------------------------------------------------------------------------

typedef __bf16 bf16_t;
typedef __attribute__((ext_vector_type(16))) __bf16 v16bf;
typedef __attribute__((ext_vector_type(8)))  __bf16 v8bf;
typedef __attribute__((ext_vector_type(8)))  float  v8f;

#define N_IN    4096
#define N_MID   4096
#define N_OUT   2
#define T_STEPS 20
#define BATCH   128
#define KSPLIT  8
#define L_TAU   0.8f

// ---------------- elementwise prep kernels ----------------

__global__ void k_cvt_bf16(const float* __restrict__ src, bf16_t* __restrict__ dst, int n) {
    int i = blockIdx.x * blockDim.x + threadIdx.x;
    int stride = gridDim.x * blockDim.x;
    for (; i < n; i += stride) dst[i] = (bf16_t)src[i];
}

// x: [B, N_IN, T] fp32  ->  xbf: [T, B, N_IN] bf16
__global__ void k_prep_x(const float* __restrict__ x, bf16_t* __restrict__ xbf) {
    int idx = blockIdx.x * blockDim.x + threadIdx.x;   // over B*N_IN
    if (idx >= BATCH * N_IN) return;
    const float* src = x + (size_t)idx * T_STEPS;      // idx = b*N_IN + i, contiguous T run
    #pragma unroll
    for (int t = 0; t < T_STEPS; ++t)
        xbf[(size_t)t * BATCH * N_IN + idx] = (bf16_t)src[t];
}

__global__ void k_zero(float* __restrict__ p, int n) {
    int i = blockIdx.x * blockDim.x + threadIdx.x;
    int stride = gridDim.x * blockDim.x;
    for (; i < n; i += stride) p[i] = 0.f;
}

// zero s4/y4 (512 floats) and out_rec[:, 0, :] (the initial all-zero output)
__global__ void k_init_small(float* __restrict__ s4y4, float* __restrict__ out) {
    int t = threadIdx.x;                 // 512 threads
    if (t < 512) s4y4[t] = 0.f;
    if (t < 256) out[257 + (t >> 1) * 42 + (t & 1)] = 0.f;
}

// ---------------- main WMMA GEMM (split-K partial) ----------------
// out_partial[kc][m][n] = sum_{k in chunk kc} act[m,k] * W[n,k]
// grid = (N_MID/128, KSPLIT), block = 128 threads (4 waves, 64x64 tile each)

__global__ __launch_bounds__(128)
void k_snu_gemm(const bf16_t* __restrict__ act,   // [BATCH, 4096] bf16 row-major
                const bf16_t* __restrict__ W,     // [4096, 4096] bf16 row-major (out,in)
                float* __restrict__ part)         // [KSPLIT, BATCH, N_MID]
{
    const int lane  = threadIdx.x & 31;
    const int wave  = threadIdx.x >> 5;       // 0..3
    const int m0    = (wave >> 1) * 64;       // wave row base (M=128 total)
    const int n0    = blockIdx.x * 128 + (wave & 1) * 64;
    const int kc    = blockIdx.y;
    const int kbeg  = kc * (N_MID / KSPLIT);  // 512-wide K chunk
    const int kend  = kbeg + (N_MID / KSPLIT);

    const int rA = lane & 15;                 // row-in-tile for both A and B frags
    const int hA = lane >> 4;                 // lane half selects K sub-range

    v8f acc[4][4] = {};                       // 64x64 f32 accumulator tile

    union Frag { v16bf v; struct { v8bf lo, hi; } p; };

    for (int kb = kbeg; kb < kend; kb += 32) {
        Frag a[4], b[4];
        // A 16x32 bf16 layout: lane (m=rA, half hA) holds k = [16h .. 16h+15] contiguous
        #pragma unroll
        for (int mi = 0; mi < 4; ++mi) {
            const bf16_t* pa = act + (size_t)(m0 + mi * 16 + rA) * N_IN + kb + hA * 16;
            a[mi].p.lo = *(const v8bf*)(pa);
            a[mi].p.hi = *(const v8bf*)(pa + 8);
        }
        // B 32x16 bf16 layout: lane (n=rA, half hA) holds k = [16h .. 16h+15] contiguous.
        // B[k,n] = W[n,k] => contiguous 32B run of W row (n0+ni*16+rA).
        #pragma unroll
        for (int ni = 0; ni < 4; ++ni) {
            const bf16_t* pb = W + (size_t)(n0 + ni * 16 + rA) * N_IN + kb + hA * 16;
            b[ni].v = *(const v16bf*)(pb);
        }
        #pragma unroll
        for (int mi = 0; mi < 4; ++mi)
            #pragma unroll
            for (int ni = 0; ni < 4; ++ni)
                acc[mi][ni] = __builtin_amdgcn_wmma_f32_16x16x32_bf16(
                    false, a[mi].v, false, b[ni].v,
                    (short)0, acc[mi][ni], false, false);
    }

    // C/D layout: VGPR j of lane L -> (m = j + 8*(L>>4), n = L&15)
    float* base = part + (size_t)kc * BATCH * N_MID;
    const int rowHalf = hA * 8;
    const int col     = rA;
    #pragma unroll
    for (int mi = 0; mi < 4; ++mi)
        #pragma unroll
        for (int ni = 0; ni < 4; ++ni) {
            const int n = n0 + ni * 16 + col;
            #pragma unroll
            for (int j = 0; j < 8; ++j) {
                const int m = m0 + mi * 16 + rowHalf + j;
                base[(size_t)m * N_MID + n] = acc[mi][ni][j];
            }
        }
}

// ---------------- split-K reduce + fused SNU update ----------------
// s_new = relu(gemm + l_tau * s * (1-y));  y_new = sigmoid(s_new + b)

__global__ __launch_bounds__(256)
void k_snu_update(const float* __restrict__ part,   // [KSPLIT, BATCH, N_MID]
                  const float* __restrict__ bias,   // [N_MID]
                  float* __restrict__ s,            // [BATCH, N_MID] state (in/out)
                  float* __restrict__ y,            // [BATCH, N_MID] state (in/out)
                  bf16_t* __restrict__ ybf)         // [BATCH, N_MID] next-layer activation
{
    int idx = blockIdx.x * blockDim.x + threadIdx.x;
    if (idx >= BATCH * N_MID) return;
    float g = 0.f;
    #pragma unroll
    for (int kc = 0; kc < KSPLIT; ++kc)
        g += part[(size_t)kc * BATCH * N_MID + idx];
    const int n = idx & (N_MID - 1);
    const float s_old = s[idx];
    const float y_old = y[idx];
    float s_new = fmaxf(g + L_TAU * s_old * (1.f - y_old), 0.f);
    float y_new = 1.f / (1.f + __expf(-(s_new + bias[n])));
    s[idx] = s_new;
    y[idx] = y_new;
    ybf[idx] = (bf16_t)y_new;
}

// ---------------- tiny output layer (N_OUT = 2) ----------------

__global__ __launch_bounds__(256)
void k_layer4(const float* __restrict__ y3,   // [BATCH, N_MID] fp32
              const float* __restrict__ W4,   // [2, N_MID] fp32
              const float* __restrict__ b4,   // [2]
              float* __restrict__ s4,         // [BATCH*2]
              float* __restrict__ y4,         // [BATCH*2]
              float* __restrict__ out,        // d_out base
              int t)
{
    __shared__ float red0[256], red1[256];
    const int b   = blockIdx.x;
    const int tid = threadIdx.x;
    float p0 = 0.f, p1 = 0.f;
    for (int i = tid; i < N_MID; i += 256) {
        const float v = y3[(size_t)b * N_MID + i];
        p0 += v * W4[i];
        p1 += v * W4[N_MID + i];
    }
    red0[tid] = p0; red1[tid] = p1;
    __syncthreads();
    for (int st = 128; st > 0; st >>= 1) {
        if (tid < st) { red0[tid] += red0[tid + st]; red1[tid] += red1[tid + st]; }
        __syncthreads();
    }
    if (tid == 0) {
        #pragma unroll
        for (int o = 0; o < 2; ++o) {
            const float g  = o ? red1[0] : red0[0];
            const int   si = b * 2 + o;
            const float sn = fmaxf(g + L_TAU * s4[si] * (1.f - y4[si]), 0.f);
            const float yn = 1.f / (1.f + __expf(-(sn + b4[o])));
            s4[si] = sn; y4[si] = yn;
            // out_rec[b][t+1][o] at offset 257 + b*42 + (t+1)*2 + o
            out[257 + b * 42 + (t + 1) * 2 + o] = yn;
        }
    }
}

// ---------------- loss / accuracy ----------------

__global__ __launch_bounds__(128)
void k_final(const int* __restrict__ y, float* __restrict__ out)
{
    __shared__ float lred[128], ared[128];
    const int b = threadIdx.x;             // 128 threads, one per batch row
    const float* rec = out + 257 + b * 42; // out_rec[b][:][:], 21 time slices
    float m0 = 0.f, m1 = 0.f;
    #pragma unroll
    for (int t = 0; t <= T_STEPS; ++t) { m0 += rec[t * 2]; m1 += rec[t * 2 + 1]; }
    m0 *= (1.f / 20.f); m1 *= (1.f / 20.f);
    out[1 + b * 2]     = m0;
    out[1 + b * 2 + 1] = m1;
    const int label = y[b];
    const float mx  = fmaxf(m0, m1);
    const float lse = mx + __logf(__expf(m0 - mx) + __expf(m1 - mx));
    const float logp = (label == 0 ? m0 : m1) - lse;
    const int pred = (m1 > m0) ? 1 : 0;    // argmax, first-index tie-break
    lred[b] = -logp;
    ared[b] = (pred == label) ? 1.f : 0.f;
    __syncthreads();
    for (int st = 64; st > 0; st >>= 1) {
        if (b < st) { lred[b] += lred[b + st]; ared[b] += ared[b + st]; }
        __syncthreads();
    }
    if (b == 0) {
        out[0]    = lred[0] * (1.f / 128.f);  // loss
        out[5633] = ared[0] * (1.f / 128.f);  // acc
    }
}

// ---------------- host launcher ----------------

extern "C" void kernel_launch(void* const* d_in, const int* in_sizes, int n_in,
                              void* d_out, int out_size, void* d_ws, size_t ws_size,
                              hipStream_t stream)
{
    (void)in_sizes; (void)n_in; (void)out_size; (void)ws_size;
    const float* x  = (const float*)d_in[0];
    const int*   y  = (const int*)  d_in[1];
    const float* W1 = (const float*)d_in[2];
    const float* b1 = (const float*)d_in[3];
    const float* W2 = (const float*)d_in[4];
    const float* b2 = (const float*)d_in[5];
    const float* W3 = (const float*)d_in[6];
    const float* b3 = (const float*)d_in[7];
    const float* W4 = (const float*)d_in[8];
    const float* b4 = (const float*)d_in[9];
    float* out = (float*)d_out;

    // ---- workspace carve-up (256B aligned) ----
    char*  ws  = (char*)d_ws;
    size_t off = 0;
    auto alloc = [&](size_t bytes) -> void* {
        void* p = ws + off;
        off = (off + bytes + 255) & ~(size_t)255;
        return p;
    };
    const size_t NW    = (size_t)N_MID * N_MID;     // 16M elements per weight
    const size_t NA    = (size_t)BATCH * N_MID;     // 512K activation elements
    bf16_t* Wbf   = (bf16_t*)alloc(3 * NW * sizeof(bf16_t));            // 96 MB
    bf16_t* xbf   = (bf16_t*)alloc((size_t)T_STEPS * NA * sizeof(bf16_t)); // 20 MB
    bf16_t* actbf = (bf16_t*)alloc(3 * NA * sizeof(bf16_t));            // y1/y2/y3 bf16
    float*  sAll  = (float*) alloc(3 * NA * sizeof(float));             // s1/s2/s3
    float*  yAll  = (float*) alloc(3 * NA * sizeof(float));             // y1/y2/y3
    float*  s4y4  = (float*) alloc(512 * sizeof(float));                // s4|y4
    float*  part  = (float*) alloc((size_t)KSPLIT * NA * sizeof(float)); // 16 MB

    // ---- one-time (per call) prep: bf16 weights, x transpose, zero state ----
    k_cvt_bf16<<<8192, 256, 0, stream>>>(W1, Wbf,           (int)NW);
    k_cvt_bf16<<<8192, 256, 0, stream>>>(W2, Wbf + NW,      (int)NW);
    k_cvt_bf16<<<8192, 256, 0, stream>>>(W3, Wbf + 2 * NW,  (int)NW);
    k_prep_x  <<<(BATCH * N_IN) / 256, 256, 0, stream>>>(x, xbf);
    k_zero    <<<2048, 256, 0, stream>>>(sAll, (int)(3 * NA));
    k_zero    <<<2048, 256, 0, stream>>>(yAll, (int)(3 * NA));
    k_init_small<<<1, 512, 0, stream>>>(s4y4, out);

    // ---- time recurrence: 3 WMMA GEMMs + tiny layer per step ----
    const dim3 ggrid(N_MID / 128, KSPLIT);
    const int  ublocks = (int)(NA / 256);
    for (int t = 0; t < T_STEPS; ++t) {
        const bf16_t* in1 = xbf + (size_t)t * NA;
        // layer 1
        k_snu_gemm  <<<ggrid, 128, 0, stream>>>(in1, Wbf, part);
        k_snu_update<<<ublocks, 256, 0, stream>>>(part, b1, sAll, yAll, actbf);
        // layer 2
        k_snu_gemm  <<<ggrid, 128, 0, stream>>>(actbf, Wbf + NW, part);
        k_snu_update<<<ublocks, 256, 0, stream>>>(part, b2, sAll + NA, yAll + NA, actbf + NA);
        // layer 3
        k_snu_gemm  <<<ggrid, 128, 0, stream>>>(actbf + NA, Wbf + 2 * NW, part);
        k_snu_update<<<ublocks, 256, 0, stream>>>(part, b3, sAll + 2 * NA, yAll + 2 * NA, actbf + 2 * NA);
        // layer 4 (N_OUT = 2)
        k_layer4<<<BATCH, 256, 0, stream>>>(yAll + 2 * NA, W4, b4, s4y4, s4y4 + 256, out, t);
    }

    // ---- loss / m / acc ----
    k_final<<<1, 128, 0, stream>>>(y, out);
}
